// GCN_54339926229596
// MI455X (gfx1250) — matrix operand order repaired and checked by reading, not compiled, not run on hardware
//
#include <hip/hip_runtime.h>
#include <hip/hip_bf16.h>

// ---------------------------------------------------------------------------
// Dead-code-eliminated GCN: the 72-node graph reduces (2-hop cone of node 0)
// to   g = carry + x[:,i1] + x[:,i3] + x[:,i5]
//      u = relu(W1 g + b1);  carry' = relu(W2 u + b2)
// Batched 16 samples/block so each step is a 16x128 @ 128x128 GEMM done with
// v_wmma_f32_16x16x32_f16 (f32 accumulate). Memory-bound on the column gather.
// ---------------------------------------------------------------------------

typedef __attribute__((ext_vector_type(16))) _Float16 v16h;
typedef __attribute__((ext_vector_type(8)))  _Float16 v8h;
typedef __attribute__((ext_vector_type(8)))  float    v8f;

#define DIM     128
#define NSTEPS  10
#define NCHUNK  71
#define SPB     16          // samples per block (M dimension of WMMA)
#define GPITCH  (DIM + 8)   // f16 LDS row pitch (bank-conflict padding)

union H16 { v16h v; v8h h[2]; };

// A-matrix 16x32 f16 fragment (ISA 7.12.2): lane holds row M=lane%16;
// lanes 0-15: K = {kt*32+0..7, kt*32+16..23}; lanes 16-31: +8.
__device__ __forceinline__ v16h load_frag_A(const _Float16* __restrict__ row,
                                            int kt, int lane) {
    H16 u;
    const int a0 = (lane < 16) ? 0 : 8;
    const _Float16* p = row + kt * 32;
    u.h[0] = *(const v8h*)(p + a0);
    u.h[1] = *(const v8h*)(p + 16 + a0);
    return u.v;
}

// B-matrix 32x16 f16 fragment: lane holds column N=lane%16;
// lanes 0-15 -> K = {kt*32+0..7, kt*32+16..23}; lanes 16-31: +8.
// Since B = W^T (B[k,n] = W[n,k]) this reads row n of W directly.
__device__ __forceinline__ v16h load_frag_B(const _Float16* __restrict__ W,
                                            int ncol, int kt, int lane) {
    H16 u;
    const int base = (lane < 16) ? 0 : 8;
    const _Float16* p = W + ncol * DIM + kt * 32 + base;
    u.h[0] = *(const v8h*)(p);
    u.h[1] = *(const v8h*)(p + 16);
    return u.v;
}

__global__ __launch_bounds__(256) void prep_w16(const float* __restrict__ W1,
                                                const float* __restrict__ W2,
                                                _Float16* __restrict__ W16) {
    int i = blockIdx.x * 256 + threadIdx.x;
    if (i < DIM * DIM) {
        W16[i]             = (_Float16)W1[i];
        W16[i + DIM * DIM] = (_Float16)W2[i];
    }
}

__global__ __launch_bounds__(256) void gcn_collapsed(
    const float* __restrict__ x,     // (B, D, L)
    const int*   __restrict__ ii,    // (B, 710)
    const _Float16* __restrict__ W16,// W1 f16 then W2 f16, row-major 128x128
    const float* __restrict__ b1,
    const float* __restrict__ b2,
    const float* __restrict__ Wc,    // (128,)
    const float* __restrict__ bc,    // (1,)
    float* __restrict__ out,         // (B,)
    int L)
{
    __shared__ float    carry[SPB][DIM];
    __shared__ _Float16 gbuf[SPB][GPITCH];
    __shared__ _Float16 ubuf[SPB][GPITCH];

    const int t    = threadIdx.x;
    const int lane = t & 31;
    const int wave = t >> 5;              // 0..7 -> N tile
    const int bbase = blockIdx.x * SPB;

    // zero the carry
    for (int i = t; i < SPB * DIM; i += 256) ((float*)carry)[i] = 0.f;

    // resident B fragments: 4 K-tiles per layer
    const int ncol = wave * 16 + (lane & 15);
    v16h w1f[4], w2f[4];
#pragma unroll
    for (int kt = 0; kt < 4; ++kt) {
        w1f[kt] = load_frag_B(W16,             ncol, kt, lane);
        w2f[kt] = load_frag_B(W16 + DIM * DIM, ncol, kt, lane);
    }
    const float bias1 = b1[ncol];
    const float bias2 = b2[ncol];

    // gather mapping: thread t -> sample gs, features gd..gd+7
    const int gs = t >> 4;
    const int gd = (t & 15) * 8;
    const int b  = bbase + gs;
    const float* xb = x + (size_t)b * DIM * L;

    __syncthreads();

    for (int step = 0; step < NSTEPS; ++step) {
        // ---- build g = carry + x[:,i1] + x[:,i3] + x[:,i5] (only live rows)
        {
            const int ibo = b * (NCHUNK * NSTEPS) + step * NCHUNK;
            const int c1 = ii[ibo + 1];
            const int c3 = ii[ibo + 3];
            const int c5 = ii[ibo + 5];
#pragma unroll
            for (int q = 0; q < 8; ++q) {
                const int d = gd + q;
                const size_t dl = (size_t)d * L;
                float v = carry[gs][d] + xb[dl + c1] + xb[dl + c3] + xb[dl + c5];
                gbuf[gs][d] = (_Float16)v;
            }
        }
        __syncthreads();

        // ---- u = relu(W1 g + b1) : 16x128 @ 128x128 via WMMA
        {
            v8f acc = {};
#pragma unroll
            for (int kt = 0; kt < 4; ++kt) {
                v16h a = load_frag_A(&gbuf[lane & 15][0], kt, lane);
                acc = __builtin_amdgcn_wmma_f32_16x16x32_f16(
                          false, a, false, w1f[kt], (short)0, acc, false, false);
            }
#pragma unroll
            for (int r = 0; r < 8; ++r) {
                const int s = (lane < 16) ? r : r + 8;   // C/D layout: VGPR r
                float v = acc[r] + bias1;
                ubuf[s][ncol] = (_Float16)(v > 0.f ? v : 0.f);
            }
        }
        __syncthreads();

        // ---- carry' = relu(W2 u + b2)
        {
            v8f acc = {};
#pragma unroll
            for (int kt = 0; kt < 4; ++kt) {
                v16h a = load_frag_A(&ubuf[lane & 15][0], kt, lane);
                acc = __builtin_amdgcn_wmma_f32_16x16x32_f16(
                          false, a, false, w2f[kt], (short)0, acc, false, false);
            }
#pragma unroll
            for (int r = 0; r < 8; ++r) {
                const int s = (lane < 16) ? r : r + 8;
                float v = acc[r] + bias2;
                carry[s][ncol] = (v > 0.f ? v : 0.f);
            }
        }
        __syncthreads();
    }

    // ---- out = Wc . carry + bc
    if (t < SPB) {
        float acc = bc[0];
#pragma unroll 8
        for (int d = 0; d < DIM; ++d) acc += carry[t][d] * Wc[d];
        out[bbase + t] = acc;
    }
}

extern "C" void kernel_launch(void* const* d_in, const int* in_sizes, int n_in,
                              void* d_out, int out_size, void* d_ws, size_t ws_size,
                              hipStream_t stream) {
    const float* x  = (const float*)d_in[0];
    const int*   ii = (const int*)  d_in[1];
    const float* W1 = (const float*)d_in[2];
    const float* b1 = (const float*)d_in[3];
    const float* W2 = (const float*)d_in[4];
    const float* b2 = (const float*)d_in[5];
    const float* Wc = (const float*)d_in[6];
    const float* bc = (const float*)d_in[7];
    float* out = (float*)d_out;

    _Float16* W16 = (_Float16*)d_ws;            // 2 * 128*128 f16 = 64 KB

    const int B = out_size;                     // 1024 samples
    const int L = in_sizes[0] / (B * DIM);      // 1024

    prep_w16<<<(DIM * DIM + 255) / 256, 256, 0, stream>>>(W1, W2, W16);
    gcn_collapsed<<<B / SPB, 256, 0, stream>>>(x, ii, W16, b1, b2, Wc, bc, out, L);
}
